// SelfAttention_14164802142349
// MI455X (gfx1250) — compile-verified, matrix operand order checked by reading, not run on hardware
//
#include <hip/hip_runtime.h>
#include <hip/hip_bf16.h>

// Problem constants (match reference)
#define B_ 4
#define S_ 2048
#define E_ 1024
#define H_ 16
#define D_ 64
static constexpr float SCALE_ = 0.125f; // D^-0.5

typedef __attribute__((ext_vector_type(16))) __bf16 v16bf;
typedef __attribute__((ext_vector_type(8)))  __bf16 v8bf;
typedef __attribute__((ext_vector_type(8)))  float  v8f;
typedef __attribute__((ext_vector_type(4)))  float  v4f;
typedef __attribute__((ext_vector_type(4)))  int    v4i;

__device__ __forceinline__ v8f vzero8() {
  v8f z;
#pragma unroll
  for (int i = 0; i < 8; ++i) z[i] = 0.0f;
  return z;
}

__device__ __forceinline__ v8f bwmma(v16bf a, v16bf b, v8f c) {
  // D = A(16x32 bf16) * B(32x16 bf16) + C(16x16 f32)
  return __builtin_amdgcn_wmma_f32_16x16x32_bf16(
      /*neg_a=*/false, a, /*neg_b=*/false, b,
      /*c_mod=*/(short)0, c, /*reuse_a=*/false, /*reuse_b=*/false);
}

// Async 16-byte global -> LDS copy (gfx1250 GLOBAL_LOAD_ASYNC_TO_LDS_B128,
// tracked on ASYNCcnt). Builtin signature (from compiler diagnostics):
//   void (v4i addrspace(1)*, v4i addrspace(3)*, imm int offset, imm int cpol)
__device__ __forceinline__ void async_copy16(const __bf16* g, __bf16* l) {
#if __has_builtin(__builtin_amdgcn_global_load_async_to_lds_b128)
  typedef __attribute__((address_space(1))) v4i* gv4i_t;
  typedef __attribute__((address_space(3))) v4i* lv4i_t;
  __builtin_amdgcn_global_load_async_to_lds_b128(
      (gv4i_t)(void*)g, (lv4i_t)(void*)l, /*offset=*/0, /*cpol=*/0);
#else
  *(v8bf*)l = *(const v8bf*)g;
#endif
}

__device__ __forceinline__ void async_wait0() {
#if __has_builtin(__builtin_amdgcn_s_wait_asynccnt)
  __builtin_amdgcn_s_wait_asynccnt(0);
#elif __has_builtin(__builtin_amdgcn_global_load_async_to_lds_b128)
  asm volatile("s_wait_asynccnt 0x0" ::: "memory");
#endif
}

// 16x32 bf16 WMMA operand fragment from a row-major bf16 matrix (global or
// LDS). Lane holds one source row (row0 + lane%16); K striped:
// vgprs 0-3 <- K = half*8 + 0..7, vgprs 4-7 <- K = 16 + half*8 + 0..7.
__device__ __forceinline__ v16bf load_frag_bf16(const __bf16* base,
                                                int ld, int row0, int k0) {
  const int lane = threadIdx.x & 31;
  const int half = lane >> 4;
  const int r    = lane & 15;
  const __bf16* p = base + (size_t)(row0 + r) * ld + k0 + half * 8;
  union U { v16bf v; v8bf h[2]; } u;
  u.h[0] = *(const v8bf*)(p);
  u.h[1] = *(const v8bf*)(p + 16);
  return u.v;
}

// ---------------------------------------------------------------------------
// Pre-pass: fp32 -> bf16 elementwise (n must be a multiple of 8; it is).
// One-time conversion so the GEMM inner loops are pure b128-load -> wmma.
// ---------------------------------------------------------------------------
__global__ __launch_bounds__(256) void cvt_f32_bf16_kernel(
    const float* __restrict__ src, __bf16* __restrict__ dst, size_t n) {
  size_t i = ((size_t)blockIdx.x * blockDim.x + threadIdx.x) * 8;
  if (i + 8 > n) return;
  v4f a = *(const v4f*)(src + i);
  v4f b = *(const v4f*)(src + i + 4);
  v8bf o;
#pragma unroll
  for (int j = 0; j < 4; ++j) {
    o[j]     = (__bf16)a[j];
    o[4 + j] = (__bf16)b[j];
  }
  *(v8bf*)(dst + i) = o;
}

// ---------------------------------------------------------------------------
// Kernel 1: proj = x @ W^T + bias, fused RoPE on Q/K, scatter to
//   Q,K : [B,H,S,D] bf16      V : [B,H,D,S] bf16 (transposed for P@V operand)
// Block = 4 waves; wave computes 32(M) x 64(N); every B fragment feeds 2 WMMAs.
// ---------------------------------------------------------------------------
__global__ __launch_bounds__(128) void qkv_rope_kernel(
    const __bf16* __restrict__ xb, const __bf16* __restrict__ wb,
    const float* __restrict__ bias,
    const float* __restrict__ cosT, const float* __restrict__ sinT,
    __bf16* __restrict__ qb, __bf16* __restrict__ kb, __bf16* __restrict__ vt) {
  const int wave = threadIdx.x >> 5;
  const int lane = threadIdx.x & 31;
  const int half = lane >> 4;
  const int c    = lane & 15;
  const int m0   = blockIdx.x * 128 + wave * 32;  // row in [B*S)
  const int n0   = blockIdx.y * 64;               // col in [3E)

  v8f acc[2][4];
#pragma unroll
  for (int ms = 0; ms < 2; ++ms)
#pragma unroll
    for (int t = 0; t < 4; ++t) acc[ms][t] = vzero8();

  for (int k = 0; k < E_; k += 32) {
    v16bf a0 = load_frag_bf16(xb, E_, m0, k);
    v16bf a1 = load_frag_bf16(xb, E_, m0 + 16, k);
#pragma unroll
    for (int t = 0; t < 4; ++t) {
      v16bf bfr = load_frag_bf16(wb, E_, n0 + t * 16, k);
      acc[0][t] = bwmma(a0, bfr, acc[0][t]);
      acc[1][t] = bwmma(a1, bfr, acc[1][t]);
    }
  }

  // bias (per-column, constant over rows)
#pragma unroll
  for (int t = 0; t < 4; ++t) {
    float bv = bias[n0 + t * 16 + c];
#pragma unroll
    for (int ms = 0; ms < 2; ++ms)
#pragma unroll
      for (int r = 0; r < 8; ++r) acc[ms][t][r] += bv;
  }

  const int region = n0 / E_;        // 0=Q 1=K 2=V
  const int h      = (n0 % E_) / D_; // head (wave tile is head-aligned)

  if (region < 2) {
    // RoPE: rot[d] = d<32 ? -v[d+32] : v[d-32]; pairs live in tiles t and t^2.
    __bf16* dst = (region == 0) ? qb : kb;
#pragma unroll
    for (int ms = 0; ms < 2; ++ms) {
      v8f outv[4];
#pragma unroll
      for (int t = 0; t < 4; ++t) {
        const int d = t * 16 + c;
#pragma unroll
        for (int r = 0; r < 8; ++r) {
          int m = m0 + ms * 16 + r + 8 * half;
          int s = m & (S_ - 1);
          float cv  = cosT[s * D_ + d];
          float sv  = sinT[s * D_ + d];
          float rot = (t < 2) ? -acc[ms][t + 2][r] : acc[ms][t - 2][r];
          outv[t][r] = acc[ms][t][r] * cv + rot * sv;
        }
      }
#pragma unroll
      for (int t = 0; t < 4; ++t) {
        const int d = t * 16 + c;
#pragma unroll
        for (int r = 0; r < 8; ++r) {
          int m = m0 + ms * 16 + r + 8 * half;
          int b = m >> 11;          // / S_
          int s = m & (S_ - 1);
          dst[((size_t)(b * H_ + h) * S_ + s) * D_ + d] = (__bf16)outv[t][r];
        }
      }
    }
  } else {
    // V, stored transposed [B,H,D,S]
#pragma unroll
    for (int ms = 0; ms < 2; ++ms)
#pragma unroll
      for (int t = 0; t < 4; ++t) {
        const int d = t * 16 + c;
#pragma unroll
        for (int r = 0; r < 8; ++r) {
          int m = m0 + ms * 16 + r + 8 * half;
          int b = m >> 11;
          int s = m & (S_ - 1);
          vt[((size_t)(b * H_ + h) * D_ + d) * S_ + s] = (__bf16)acc[ms][t][r];
        }
      }
  }
}

// ---------------------------------------------------------------------------
// Kernel 2: flash attention per (b,h). Block = 8 waves x 32 q-rows = 256 q.
// k-tiles of 64 staged in LDS via async global->LDS DMA (ASYNCcnt path).
// Every K/V fragment is reused by two WMMAs (two q sub-tiles).
// ---------------------------------------------------------------------------
__global__ __launch_bounds__(256) void attn_kernel(
    const __bf16* __restrict__ qb, const __bf16* __restrict__ kb,
    const __bf16* __restrict__ vt, __bf16* __restrict__ ob) {
  constexpr int LDK = 72;  // padded LDS row stride (elements)
  __shared__ __align__(16) __bf16 Kt[64 * LDK];      // [k_local][d]
  __shared__ __align__(16) __bf16 Vs[64 * LDK];      // [d][k_local]
  __shared__ __align__(16) __bf16 Pb[8 * 32 * LDK];  // per-wave 32x64 P tile

  const int wave = threadIdx.x >> 5;
  const int lane = threadIdx.x & 31;
  const int half = lane >> 4;
  const int c    = lane & 15;
  const int bh   = blockIdx.y;
  const int b    = bh >> 4;   // / H_
  const int h    = bh & 15;
  const int m0   = blockIdx.x * 256 + wave * 32;  // q row within (b,h)

  const __bf16* Qbh = qb + (size_t)bh * S_ * D_;
  const __bf16* Kbh = kb + (size_t)bh * S_ * D_;
  const __bf16* Vbh = vt + (size_t)bh * D_ * S_;

  v16bf qf[2][2];  // [q subtile][kc]
#pragma unroll
  for (int qs = 0; qs < 2; ++qs)
#pragma unroll
    for (int kc = 0; kc < 2; ++kc)
      qf[qs][kc] = load_frag_bf16(Qbh, D_, m0 + qs * 16, kc * 32);

  v8f acc_o[2][4];
  float mrun[2][8], lrun[2][8];
#pragma unroll
  for (int qs = 0; qs < 2; ++qs) {
#pragma unroll
    for (int t = 0; t < 4; ++t) acc_o[qs][t] = vzero8();
#pragma unroll
    for (int r = 0; r < 8; ++r) {
      mrun[qs][r] = -__builtin_inff();
      lrun[qs][r] = 0.0f;
    }
  }

  __bf16* Pw = Pb + wave * 32 * LDK;

  for (int k0 = 0; k0 < S_; k0 += 64) {
    __syncthreads();  // previous tile fully consumed
    // Stage K (row-major [k][d]) and V^T (row-major [d][k]) tiles with
    // async global->LDS B128 copies; 2 chunks per thread per buffer.
    for (int ch = threadIdx.x; ch < 512; ch += 256) {
      int row = ch >> 3;
      int co  = (ch & 7) * 8;
      async_copy16(&Kbh[(size_t)(k0 + row) * D_ + co], &Kt[row * LDK + co]);
      async_copy16(&Vbh[(size_t)row * S_ + k0 + co],   &Vs[row * LDK + co]);
    }
    async_wait0();    // this wave's async copies landed in LDS
    __syncthreads();  // all waves' copies visible

    // scores: 32(q) x 64(k), K-dim = d = 64; each K fragment feeds 2 wmma
    v8f sc[2][4];
#pragma unroll
    for (int qs = 0; qs < 2; ++qs)
#pragma unroll
      for (int t = 0; t < 4; ++t) sc[qs][t] = vzero8();
#pragma unroll
    for (int kc = 0; kc < 2; ++kc)
#pragma unroll
      for (int t = 0; t < 4; ++t) {
        v16bf bfr = load_frag_bf16(Kt, LDK, t * 16, kc * 32);
        sc[0][t] = bwmma(qf[0][kc], bfr, sc[0][t]);
        sc[1][t] = bwmma(qf[1][kc], bfr, sc[1][t]);
      }

    // online softmax per q sub-tile
#pragma unroll
    for (int qs = 0; qs < 2; ++qs) {
#pragma unroll
      for (int t = 0; t < 4; ++t)
#pragma unroll
        for (int r = 0; r < 8; ++r) sc[qs][t][r] *= SCALE_;

      float mx[8];
#pragma unroll
      for (int r = 0; r < 8; ++r)
        mx[r] = fmaxf(fmaxf(sc[qs][0][r], sc[qs][1][r]),
                      fmaxf(sc[qs][2][r], sc[qs][3][r]));
#pragma unroll
      for (int off = 8; off >= 1; off >>= 1)
#pragma unroll
        for (int r = 0; r < 8; ++r)
          mx[r] = fmaxf(mx[r], __shfl_xor(mx[r], off, 32));

      float alpha[8];
#pragma unroll
      for (int r = 0; r < 8; ++r) {
        float mn = fmaxf(mrun[qs][r], mx[r]);
        alpha[r] = __expf(mrun[qs][r] - mn);
        mrun[qs][r] = mn;
      }

      float rs[8];
#pragma unroll
      for (int r = 0; r < 8; ++r) rs[r] = 0.0f;
#pragma unroll
      for (int t = 0; t < 4; ++t)
#pragma unroll
        for (int r = 0; r < 8; ++r) {
          float p = __expf(sc[qs][t][r] - mrun[qs][r]);
          sc[qs][t][r] = p;
          rs[r] += p;
        }
#pragma unroll
      for (int off = 8; off >= 1; off >>= 1)
#pragma unroll
        for (int r = 0; r < 8; ++r) rs[r] += __shfl_xor(rs[r], off, 32);
#pragma unroll
      for (int r = 0; r < 8; ++r)
        lrun[qs][r] = lrun[qs][r] * alpha[r] + rs[r];
#pragma unroll
      for (int t = 0; t < 4; ++t)
#pragma unroll
        for (int r = 0; r < 8; ++r) acc_o[qs][t][r] *= alpha[r];

      // P: accumulator layout -> LDS (per-wave region)
#pragma unroll
      for (int t = 0; t < 4; ++t)
#pragma unroll
        for (int r = 0; r < 8; ++r)
          Pw[(qs * 16 + r + 8 * half) * LDK + t * 16 + c] =
              (__bf16)sc[qs][t][r];
    }
    __syncthreads();  // P visible; also fences Kt reads before next stage

    v16bf pf[2][2];
#pragma unroll
    for (int qs = 0; qs < 2; ++qs)
#pragma unroll
      for (int kc = 0; kc < 2; ++kc)
        pf[qs][kc] = load_frag_bf16(Pw, LDK, qs * 16, kc * 32);

#pragma unroll
    for (int kc = 0; kc < 2; ++kc)
#pragma unroll
      for (int t = 0; t < 4; ++t) {
        v16bf vfr = load_frag_bf16(Vs, LDK, t * 16, kc * 32);
        acc_o[0][t] = bwmma(pf[0][kc], vfr, acc_o[0][t]);
        acc_o[1][t] = bwmma(pf[1][kc], vfr, acc_o[1][t]);
      }
  }

  // normalize and write merged-head output [B,S,E] bf16
#pragma unroll
  for (int qs = 0; qs < 2; ++qs) {
    float inv[8];
#pragma unroll
    for (int r = 0; r < 8; ++r) inv[r] = 1.0f / lrun[qs][r];
#pragma unroll
    for (int t = 0; t < 4; ++t)
#pragma unroll
      for (int r = 0; r < 8; ++r) {
        int s = m0 + qs * 16 + r + 8 * half;
        ob[((size_t)(b * S_ + s)) * E_ + h * D_ + t * 16 + c] =
            (__bf16)(acc_o[qs][t][r] * inv[r]);
      }
  }
}

// ---------------------------------------------------------------------------
// Kernel 3: out = O @ out_w^T + out_b  (fp32 result)
// Wave tile 32x64; all-bf16 operands.
// ---------------------------------------------------------------------------
__global__ __launch_bounds__(128) void outproj_kernel(
    const __bf16* __restrict__ ob, const __bf16* __restrict__ wb,
    const float* __restrict__ bias, float* __restrict__ out) {
  const int wave = threadIdx.x >> 5;
  const int lane = threadIdx.x & 31;
  const int half = lane >> 4;
  const int c    = lane & 15;
  const int m0   = blockIdx.x * 128 + wave * 32;
  const int n0   = blockIdx.y * 64;

  v8f acc[2][4];
#pragma unroll
  for (int ms = 0; ms < 2; ++ms)
#pragma unroll
    for (int t = 0; t < 4; ++t) acc[ms][t] = vzero8();

  for (int k = 0; k < E_; k += 32) {
    v16bf a0 = load_frag_bf16(ob, E_, m0, k);
    v16bf a1 = load_frag_bf16(ob, E_, m0 + 16, k);
#pragma unroll
    for (int t = 0; t < 4; ++t) {
      v16bf bfr = load_frag_bf16(wb, E_, n0 + t * 16, k);
      acc[0][t] = bwmma(a0, bfr, acc[0][t]);
      acc[1][t] = bwmma(a1, bfr, acc[1][t]);
    }
  }
#pragma unroll
  for (int t = 0; t < 4; ++t) {
    float bv = bias[n0 + t * 16 + c];
#pragma unroll
    for (int ms = 0; ms < 2; ++ms)
#pragma unroll
      for (int r = 0; r < 8; ++r) {
        int m = m0 + ms * 16 + r + 8 * half;
        out[(size_t)m * E_ + n0 + t * 16 + c] = acc[ms][t][r] + bv;
      }
  }
}

extern "C" void kernel_launch(void* const* d_in, const int* in_sizes, int n_in,
                              void* d_out, int out_size, void* d_ws, size_t ws_size,
                              hipStream_t stream) {
  (void)in_sizes; (void)n_in; (void)out_size; (void)ws_size;
  const float* x    = (const float*)d_in[0];
  const float* ipw  = (const float*)d_in[1];  // (3E, E)
  const float* ipb  = (const float*)d_in[2];  // (3E,)
  const float* ow   = (const float*)d_in[3];  // (E, E)
  const float* obia = (const float*)d_in[4];  // (E,)
  const float* cosT = (const float*)d_in[5];  // (S, D)
  const float* sinT = (const float*)d_in[6];  // (S, D)
  float* out = (float*)d_out;

  const size_t n   = (size_t)B_ * H_ * S_ * D_;  // == B*S*E = 8M elements
  const size_t nw  = (size_t)3 * E_ * E_;        // 3M elements
  const size_t now = (size_t)E_ * E_;            // 1M elements
  __bf16* qb   = (__bf16*)d_ws;
  __bf16* kbuf = qb + n;
  __bf16* vt   = kbuf + n;
  __bf16* obuf = vt + n;
  __bf16* xb   = obuf + n;   // x in bf16
  __bf16* wb   = xb + n;     // in_proj_weight in bf16
  __bf16* owb  = wb + nw;    // out_w in bf16;  total scratch = 88 MB

  // one-time fp32 -> bf16 conversions (removes all cvt work from GEMM loops)
  cvt_f32_bf16_kernel<<<(unsigned)(n / 8 / 256), 256, 0, stream>>>(x, xb, n);
  cvt_f32_bf16_kernel<<<(unsigned)(nw / 8 / 256), 256, 0, stream>>>(ipw, wb, nw);
  cvt_f32_bf16_kernel<<<(unsigned)(now / 8 / 256), 256, 0, stream>>>(ow, owb, now);

  qkv_rope_kernel<<<dim3((B_ * S_) / 128, (3 * E_) / 64), 128, 0, stream>>>(
      xb, wb, ipb, cosT, sinT, qb, kbuf, vt);
  attn_kernel<<<dim3(S_ / 256, B_ * H_), 256, 0, stream>>>(qb, kbuf, vt, obuf);
  outproj_kernel<<<dim3((B_ * S_) / 128, E_ / 64), 128, 0, stream>>>(
      obuf, owb, obia, out);
}
